// InvariantPointMlpAttention_70987219468955
// MI455X (gfx1250) — matrix operand order repaired and checked by reading, not compiled, not run on hardware
//
#include <hip/hip_runtime.h>
#include <hip/hip_bf16.h>
#include <math.h>

#define CSd 384
#define CZd 128
#define CHd 192
#define NHD 12
#define PQn 4
#define PVn 8
#define ET  32   // edges/nodes per block tile (2 x 16 M sub-tiles)

typedef __attribute__((ext_vector_type(16))) _Float16 v16h;
typedef __attribute__((ext_vector_type(8)))  float    v8f;

// ---------------------------------------------------------------- WMMA helpers

__device__ __forceinline__ v8f wmma_f16(v16h a, v16h b, v8f c) {
  return __builtin_amdgcn_wmma_f32_16x16x32_f16(false, a, false, b, (short)0, c,
                                                false, false);
}

// A fragment (16x32 f16, M x K) from LDS row-major [16][ld]:
// lanes 0-15: row=lane, K = kc+0..7 and kc+16..23
// lanes 16-31: row=lane-16, K = kc+8..15 and kc+24..31
__device__ __forceinline__ v16h lds_loadA(const _Float16* x, int lane, int kc, int ld) {
  int row = lane & 15;
  int hi = (lane >> 4) & 1;
  const _Float16* p = x + row * ld + kc + hi * 8;
  v16h a;
  ((uint4*)&a)[0] = *(const uint4*)(p);
  ((uint4*)&a)[1] = *(const uint4*)(p + 16);
  return a;
}

// B fragment from pre-packed global weights (tile = kt*NT + nt):
// lane n<16 holds K=0..7,16..23 of column n; lane n+16 holds K=8..15,24..31 of column n.
__device__ __forceinline__ v16h glb_loadB(const _Float16* w, int tile, int lane) {
  return *(const v16h*)(w + (((size_t)tile * 32 + lane) << 4));
}

// Software-pipelined GEMM over KT K-chunks with MB-way M register blocking.
// Next iteration's A/B fragments are issued before the current WMMAs so the
// scheduler can overlap loads with matrix ops (waits on loadcnt<=N, not 0).
template <int KT, int MB>
__device__ __forceinline__ void gemm_mb(const _Float16* A, int lda,
                                        const _Float16* B, int NT, int nt,
                                        int lane, v8f* acc) {
  v16h b = glb_loadB(B, nt, lane);
  v16h a[MB];
#pragma unroll
  for (int m = 0; m < MB; ++m) a[m] = lds_loadA(A + m * 16 * lda, lane, 0, lda);
#pragma unroll
  for (int kc = 1; kc < KT; ++kc) {
    v16h bn = glb_loadB(B, kc * NT + nt, lane);
    v16h an[MB];
#pragma unroll
    for (int m = 0; m < MB; ++m) an[m] = lds_loadA(A + m * 16 * lda, lane, kc * 32, lda);
#pragma unroll
    for (int m = 0; m < MB; ++m) acc[m] = wmma_f16(a[m], b, acc[m]);
    b = bn;
#pragma unroll
    for (int m = 0; m < MB; ++m) a[m] = an[m];
  }
#pragma unroll
  for (int m = 0; m < MB; ++m) acc[m] = wmma_f16(a[m], b, acc[m]);
}

// ---------------------------------------------------------------- pack / init

__global__ void pack_weight(const float* __restrict__ W, _Float16* __restrict__ out,
                            int K, int N, int KT, int NT) {
  size_t idx = (size_t)blockIdx.x * blockDim.x + threadIdx.x;
  size_t total = (size_t)KT * NT * 512;
  if (idx >= total) return;
  int j    = (int)(idx & 15);
  int lane = (int)((idx >> 4) & 31);
  size_t tile = idx >> 9;
  int nt = (int)(tile % NT);
  int kt = (int)(tile / NT);
  int n = nt * 16 + (lane & 15);
  int k = kt * 32 + (j < 8 ? ((lane >> 4) * 8 + j) : (16 + (lane >> 4) * 8 + (j - 8)));
  float v = (k < K && n < N) ? W[(size_t)k * N + n] : 0.f;
  out[idx] = (_Float16)v;
}

__global__ void init_ws(int* __restrict__ amax, float* __restrict__ zeros,
                        size_t nz, size_t na) {
  size_t idx = (size_t)blockIdx.x * blockDim.x + threadIdx.x;
  if (idx < na) amax[idx] = (int)0x80000000;  // encoded -inf
  if (idx < nz) zeros[idx] = 0.f;
}

// ---------------------------------------------------------------- K1: node projections (16-node tiles)

__global__ __launch_bounds__(128) void node_proj(
    const float* __restrict__ s, const float* __restrict__ rots,
    const float* __restrict__ trans,
    const _Float16* __restrict__ pWq, const float* __restrict__ bq,
    const _Float16* __restrict__ pWkv, const float* __restrict__ bkv,
    float* __restrict__ qpts, float* __restrict__ kvpts, int nN) {
  __shared__ __align__(16) _Float16 xs[16 * CSd];
  __shared__ float proj[16 * 576];  // 144 (q) + 432 (kv)
  const int tid = threadIdx.x;
  const int lane = tid & 31;
  const int wid = tid >> 5;
  const int n0 = blockIdx.x * 16;

  for (int r = wid; r < 16; r += 4) {
    int node = n0 + r;
    bool valid = node < nN;
    const float* ps = s + (size_t)(valid ? node : 0) * CSd;
    for (int c = lane; c < CSd; c += 32) xs[r * CSd + c] = (_Float16)(valid ? ps[c] : 0.f);
  }
  __syncthreads();

  const int nn = lane & 15;
  const int mb = (lane >> 4) * 8;
  for (int t = wid; t < 36; t += 4) {  // 9 Wq tiles + 27 Wkv tiles, wave-uniform split
    v8f acc = {};
    if (t < 9) {
      gemm_mb<12, 1>(xs, CSd, pWq, 9, t, lane, &acc);
      float bias = bq[t * 16 + nn];
      for (int v = 0; v < 8; ++v) proj[(mb + v) * 576 + t * 16 + nn] = acc[v] + bias;
    } else {
      int u = t - 9;
      gemm_mb<12, 1>(xs, CSd, pWkv, 27, u, lane, &acc);
      float bias = bkv[u * 16 + nn];
      for (int v = 0; v < 8; ++v) proj[(mb + v) * 576 + 144 + u * 16 + nn] = acc[v] + bias;
    }
  }
  __syncthreads();

  // rotate+translate:  out_i = sum_j R[i][j] * flat[j*P + pt] + 0.1*t_i
  for (int idx = tid; idx < 16 * 192; idx += 128) {
    int r = idx / 192, pt = idx - r * 192;
    int node = n0 + r;
    if (node >= nN) continue;
    const float* R = rots + (size_t)node * 9;
    const float* tp = trans + (size_t)node * 3;
    float t0 = tp[0] * 0.1f, t1 = tp[1] * 0.1f, t2 = tp[2] * 0.1f;
    if (pt < 48) {
      const float* fb = proj + r * 576;
      float x0 = fb[pt], x1 = fb[48 + pt], x2 = fb[96 + pt];
      float* d = qpts + (size_t)node * 144 + pt * 3;
      d[0] = R[0] * x0 + R[1] * x1 + R[2] * x2 + t0;
      d[1] = R[3] * x0 + R[4] * x1 + R[5] * x2 + t1;
      d[2] = R[6] * x0 + R[7] * x1 + R[8] * x2 + t2;
    } else {
      int pp = pt - 48;
      const float* fb = proj + r * 576 + 144;
      float x0 = fb[pp], x1 = fb[144 + pp], x2 = fb[288 + pp];
      float* d = kvpts + (size_t)node * 432 + pp * 3;
      d[0] = R[0] * x0 + R[1] * x1 + R[2] * x2 + t0;
      d[1] = R[3] * x0 + R[4] * x1 + R[5] * x2 + t1;
      d[2] = R[6] * x0 + R[7] * x1 + R[8] * x2 + t2;
    }
  }
}

// ---------------------------------------------------------------- K2: edge logits (32-edge tiles)

__global__ __launch_bounds__(128) void edge_pass1(
    const float* __restrict__ s, const float* __restrict__ z,
    const int* __restrict__ eidx, const float* __restrict__ qpts,
    const float* __restrict__ kvpts, const float* __restrict__ mask,
    const _Float16* __restrict__ pW1, const float* __restrict__ b1,
    const _Float16* __restrict__ pW2, const float* __restrict__ b2,
    const _Float16* __restrict__ pW3, const float* __restrict__ b3,
    const _Float16* __restrict__ pWb, const float* __restrict__ bbv,
    const float* __restrict__ hw,
    float* __restrict__ a_raw, int* __restrict__ amax, int nE) {
  __shared__ __align__(16) _Float16 xw[ET * 896];  // 56KB
  __shared__ __align__(16) _Float16 h1[ET * CHd];  // 12KB
  __shared__ __align__(16) _Float16 h2[ET * CHd];  // 12KB
  __shared__ float aA[ET * 16];
  __shared__ float aB[ET * 16];
  __shared__ int ssrc[ET], sdst[ET];
  const int tid = threadIdx.x;
  const int lane = tid & 31;
  const int wid = tid >> 5;
  const int e0 = blockIdx.x * ET;

  if (tid < ET) {
    int e = e0 + tid;
    int ec = e < nE ? e : 0;
    sdst[tid] = eidx[ec];       // dst = edge_index[0]
    ssrc[tid] = eidx[nE + ec];  // src = edge_index[1]
  }
  __syncthreads();
  for (int r = wid; r < ET; r += 4) {
    int e = e0 + r;
    bool valid = e < nE;
    const float* ps = s + (size_t)ssrc[r] * CSd;
    const float* pd = s + (size_t)sdst[r] * CSd;
    const float* pz = z + (size_t)(valid ? e : 0) * CZd;
    for (int c = lane; c < CSd; c += 32) {
      xw[r * 896 + c]       = (_Float16)(valid ? ps[c] : 0.f);
      xw[r * 896 + CSd + c] = (_Float16)(valid ? pd[c] : 0.f);
    }
    for (int c = lane; c < CZd; c += 32)
      xw[r * 896 + 2 * CSd + c] = (_Float16)(valid ? pz[c] : 0.f);
  }
  __syncthreads();

  const int nn = lane & 15;
  const int mb = (lane >> 4) * 8;
  for (int nt = wid; nt < 12; nt += 4) {       // layer 1: 896 -> 192, relu
    v8f acc[2] = {};
    gemm_mb<28, 2>(xw, 896, pW1, 12, nt, lane, acc);
    float bias = b1[nt * 16 + nn];
    for (int m = 0; m < 2; ++m)
      for (int v = 0; v < 8; ++v) {
        float t = acc[m][v] + bias;
        h1[(m * 16 + mb + v) * CHd + nt * 16 + nn] = (_Float16)(t > 0.f ? t : 0.f);
      }
  }
  __syncthreads();
  for (int nt = wid; nt < 12; nt += 4) {       // layer 2: 192 -> 192, relu
    v8f acc[2] = {};
    gemm_mb<6, 2>(h1, CHd, pW2, 12, nt, lane, acc);
    float bias = b2[nt * 16 + nn];
    for (int m = 0; m < 2; ++m)
      for (int v = 0; v < 8; ++v) {
        float t = acc[m][v] + bias;
        h2[(m * 16 + mb + v) * CHd + nt * 16 + nn] = (_Float16)(t > 0.f ? t : 0.f);
      }
  }
  __syncthreads();
  {  // 4 wave-uniform tasks: layer3 (192->12) on m=0/1, z@Wb (128->12) on m=0/1
    v8f acc = {};
    int m = wid & 1;
    if (wid < 2) {
      gemm_mb<6, 1>(h2 + m * 16 * CHd, CHd, pW3, 1, 0, lane, &acc);
      for (int v = 0; v < 8; ++v) aA[(m * 16 + mb + v) * 16 + nn] = acc[v];
    } else {
      gemm_mb<4, 1>(xw + 2 * CSd + m * 16 * 896, 896, pWb, 1, 0, lane, &acc);
      for (int v = 0; v < 8; ++v) aB[(m * 16 + mb + v) * 16 + nn] = acc[v];
    }
  }
  __syncthreads();

  const float c1 = 0.041666666667f;   // sqrt(1/(3*192)) = 1/24
  const float c2 = 0.5773502691896f;  // sqrt(1/3)
  for (int idx = tid; idx < ET * NHD; idx += 128) {
    int r = idx / NHD, h = idx - r * NHD;
    int e = e0 + r;
    if (e >= nE) continue;
    float sp = log1pf(expf(hw[h]));           // softplus
    float hwsp = sp * 0.13608276348796f;      // sqrt(1/(3*18)) = 1/sqrt(54)
    float aval = c1 * (aA[r * 16 + h] + b3[h]) + c2 * (aB[r * 16 + h] + bbv[h]);
    int sn = ssrc[r], dn = sdst[r];
    const float* qp = qpts + (size_t)sn * 144 + h * 12;   // head h: 4 pts x 3
    const float* kp = kvpts + (size_t)dn * 432 + h * 36;  // head h: first 4 of 12 pts
    float ssum = 0.f;
    for (int t = 0; t < 12; ++t) { float d = qp[t] - kp[t]; ssum += d * d; }
    aval += -0.5f * hwsp * ssum;
    aval += 100000.f * (mask[dn] * mask[sn] - 1.f);
    a_raw[(size_t)e * NHD + h] = aval;
    int iv = __float_as_int(aval);
    int enc = iv >= 0 ? iv : (iv ^ 0x7fffffff);  // order-preserving encoding
    atomicMax(&amax[sn * NHD + h], enc);
  }
}

// ---------------------------------------------------------------- K3: segment softmax

__global__ void edge_softmax(float* __restrict__ a_ex, const int* __restrict__ eidx,
                             const int* __restrict__ amax, float* __restrict__ denom, int nE) {
  int idx = blockIdx.x * blockDim.x + threadIdx.x;
  if (idx >= nE * NHD) return;
  int e = idx / NHD, h = idx - e * NHD;
  int sn = eidx[nE + e];
  int enc = amax[sn * NHD + h];
  float m = __int_as_float(enc >= 0 ? enc : (enc ^ 0x7fffffff));
  float ex = expf(a_ex[idx] - m);
  a_ex[idx] = ex;
  atomicAdd(&denom[sn * NHD + h], ex);
}

// ---------------------------------------------------------------- K4: v-MLP + aggregation (32-edge tiles)

__global__ __launch_bounds__(128) void edge_agg(
    const float* __restrict__ s, const float* __restrict__ z,
    const int* __restrict__ eidx, const float* __restrict__ kvpts,
    const float* __restrict__ ex, const float* __restrict__ denom,
    const _Float16* __restrict__ pV1, const float* __restrict__ b1,
    const _Float16* __restrict__ pV2, const float* __restrict__ b2,
    const _Float16* __restrict__ pV3, const float* __restrict__ b3,
    const _Float16* __restrict__ pWdz, const float* __restrict__ bdz,
    float* __restrict__ o_acc, float* __restrict__ opt_acc,
    float* __restrict__ opair_acc, int nE) {
  __shared__ __align__(16) _Float16 xv[ET * 512];  // 32KB
  __shared__ __align__(16) _Float16 h1[ET * CHd];  // 12KB
  __shared__ __align__(16) _Float16 h2[ET * CHd];  // 12KB
  __shared__ float aw[ET * NHD];
  __shared__ int ssrc[ET], sdst[ET];
  const int tid = threadIdx.x;
  const int lane = tid & 31;
  const int wid = tid >> 5;
  const int e0 = blockIdx.x * ET;

  if (tid < ET) {
    int e = e0 + tid;
    int ec = e < nE ? e : 0;
    sdst[tid] = eidx[ec];
    ssrc[tid] = eidx[nE + ec];
  }
  __syncthreads();
  for (int idx = tid; idx < ET * NHD; idx += 128) {  // normalized attention weights
    int r = idx / NHD, h = idx - r * NHD;
    int e = e0 + r;
    float w = 0.f;
    if (e < nE) w = ex[(size_t)e * NHD + h] / (denom[ssrc[r] * NHD + h] + 1e-16f);
    aw[idx] = w;
  }
  for (int r = wid; r < ET; r += 4) {  // stage [s[dst], z]
    int e = e0 + r;
    bool valid = e < nE;
    const float* pd = s + (size_t)sdst[r] * CSd;
    const float* pz = z + (size_t)(valid ? e : 0) * CZd;
    for (int c = lane; c < CSd; c += 32) xv[r * 512 + c] = (_Float16)(valid ? pd[c] : 0.f);
    for (int c = lane; c < CZd; c += 32) xv[r * 512 + CSd + c] = (_Float16)(valid ? pz[c] : 0.f);
  }
  __syncthreads();

  const int nn = lane & 15;
  const int mb = (lane >> 4) * 8;
  for (int nt = wid; nt < 12; nt += 4) {  // v layer 1: 512 -> 192, relu
    v8f acc[2] = {};
    gemm_mb<16, 2>(xv, 512, pV1, 12, nt, lane, acc);
    float bias = b1[nt * 16 + nn];
    for (int m = 0; m < 2; ++m)
      for (int v = 0; v < 8; ++v) {
        float t = acc[m][v] + bias;
        h1[(m * 16 + mb + v) * CHd + nt * 16 + nn] = (_Float16)(t > 0.f ? t : 0.f);
      }
  }
  __syncthreads();
  for (int nt = wid; nt < 12; nt += 4) {  // v layer 2: 192 -> 192, relu
    v8f acc[2] = {};
    gemm_mb<6, 2>(h1, CHd, pV2, 12, nt, lane, acc);
    float bias = b2[nt * 16 + nn];
    for (int m = 0; m < 2; ++m)
      for (int v = 0; v < 8; ++v) {
        float t = acc[m][v] + bias;
        h2[(m * 16 + mb + v) * CHd + nt * 16 + nn] = (_Float16)(t > 0.f ? t : 0.f);
      }
  }
  __syncthreads();
  for (int nt = wid; nt < 12; nt += 4) {  // v layer 3 (linear) -> o (head = nt)
    v8f acc[2] = {};
    gemm_mb<6, 2>(h2, CHd, pV3, 12, nt, lane, acc);
    float bias = b3[nt * 16 + nn];
    for (int m = 0; m < 2; ++m)
      for (int v = 0; v < 8; ++v) {
        int r = m * 16 + mb + v;
        if (e0 + r >= nE) continue;
        float w = aw[r * NHD + nt];
        atomicAdd(&o_acc[(size_t)ssrc[r] * CHd + nt * 16 + nn], w * (acc[m][v] + bias));
      }
  }
  for (int nt = wid; nt < 2; nt += 4) {  // pair_z = z @ Wdz (128 -> 32); waves 0,1
    v8f acc[2] = {};
    gemm_mb<4, 2>(xv + CSd, 512, pWdz, 2, nt, lane, acc);
    int c = nt * 16 + nn;
    float bias = bdz[c];
    for (int m = 0; m < 2; ++m)
      for (int v = 0; v < 8; ++v) {
        int r = m * 16 + mb + v;
        if (e0 + r >= nE) continue;
        float pzv = acc[m][v] + bias;
        float* dst = &opair_acc[(size_t)ssrc[r] * (NHD * 32) + c];
        for (int h = 0; h < NHD; ++h) atomicAdd(dst + h * 32, aw[r * NHD + h] * pzv);
      }
  }
  for (int idx = tid; idx < ET * NHD * PVn; idx += 128) {  // o_pt += a * v_pts[dst]
    int r = idx / (NHD * PVn);
    int rem = idx - r * (NHD * PVn);
    int h = rem / PVn, p = rem - h * PVn;
    if (e0 + r >= nE) continue;
    float w = aw[r * NHD + h];
    const float* vp = kvpts + (size_t)sdst[r] * 432 + h * 36 + (PQn + p) * 3;
    float* op = &opt_acc[(size_t)ssrc[r] * 288 + (h * PVn + p) * 3];
    atomicAdd(op + 0, w * vp[0]);
    atomicAdd(op + 1, w * vp[1]);
    atomicAdd(op + 2, w * vp[2]);
  }
}

// ---------------------------------------------------------------- K5: finalize + output GEMM (32-node tiles)

__global__ __launch_bounds__(128) void node_out(
    const float* __restrict__ o_acc, const float* __restrict__ opt_acc,
    const float* __restrict__ opair_acc, const float* __restrict__ rots,
    const float* __restrict__ trans,
    const _Float16* __restrict__ pWo, const float* __restrict__ bo,
    float* __restrict__ out, int nN) {
  __shared__ __align__(16) _Float16 feats[ET * 960];  // 60KB
  const int tid = threadIdx.x;
  const int lane = tid & 31;
  const int wid = tid >> 5;
  const int n0 = blockIdx.x * ET;

  for (int idx = tid; idx < ET * CHd; idx += 128) {  // o
    int r = idx / CHd, c = idx - r * CHd;
    int node = n0 + r;
    feats[r * 960 + c] = (_Float16)(node < nN ? o_acc[(size_t)node * CHd + c] : 0.f);
  }
  for (int idx = tid; idx < ET * 384; idx += 128) {  // o_pair
    int r = idx / 384, c = idx - r * 384;
    int node = n0 + r;
    feats[r * 960 + 576 + c] = (_Float16)(node < nN ? opair_acc[(size_t)node * 384 + c] : 0.f);
  }
  for (int idx = tid; idx < ET * 96; idx += 128) {  // inverse-rotated o_pt + norms
    int r = idx / 96, pt = idx - r * 96;
    int node = n0 + r;
    float v0 = 0.f, v1 = 0.f, v2 = 0.f, nrm = 0.f;
    if (node < nN) {
      const float* R = rots + (size_t)node * 9;
      const float* tp = trans + (size_t)node * 3;
      const float* x = opt_acc + (size_t)node * 288 + pt * 3;
      float x0 = x[0] - tp[0] * 0.1f;
      float x1 = x[1] - tp[1] * 0.1f;
      float x2 = x[2] - tp[2] * 0.1f;
      v0 = R[0] * x0 + R[3] * x1 + R[6] * x2;  // R^T (einsum 'nji,nhpj->nhpi')
      v1 = R[1] * x0 + R[4] * x1 + R[7] * x2;
      v2 = R[2] * x0 + R[5] * x1 + R[8] * x2;
      nrm = sqrtf(v0 * v0 + v1 * v1 + v2 * v2 + 1e-8f);
    }
    feats[r * 960 + 192 + pt] = (_Float16)v0;
    feats[r * 960 + 288 + pt] = (_Float16)v1;
    feats[r * 960 + 384 + pt] = (_Float16)v2;
    feats[r * 960 + 480 + pt] = (_Float16)nrm;
  }
  __syncthreads();

  const int nn = lane & 15;
  const int mb = (lane >> 4) * 8;
  for (int nt = wid; nt < 24; nt += 4) {  // feats @ Wo (960 -> 384)
    v8f acc[2] = {};
    gemm_mb<30, 2>(feats, 960, pWo, 24, nt, lane, acc);
    float bias = bo[nt * 16 + nn];
    for (int m = 0; m < 2; ++m)
      for (int v = 0; v < 8; ++v) {
        int node = n0 + m * 16 + mb + v;
        if (node < nN) out[(size_t)node * CSd + nt * 16 + nn] = acc[m][v] + bias;
      }
  }
}

// ---------------------------------------------------------------- launch

extern "C" void kernel_launch(void* const* d_in, const int* in_sizes, int n_in,
                              void* d_out, int out_size, void* d_ws, size_t ws_size,
                              hipStream_t stream) {
  (void)n_in; (void)out_size;
  const float* s     = (const float*)d_in[0];
  const float* z     = (const float*)d_in[1];
  const int*   eidx  = (const int*)d_in[2];
  const float* rots  = (const float*)d_in[3];
  const float* trans = (const float*)d_in[4];
  const float* mask  = (const float*)d_in[5];
  const float* Ww1 = (const float*)d_in[6],  *bw1 = (const float*)d_in[7];
  const float* Ww2 = (const float*)d_in[8],  *bw2 = (const float*)d_in[9];
  const float* Ww3 = (const float*)d_in[10], *bw3 = (const float*)d_in[11];
  const float* Wv1 = (const float*)d_in[12], *bv1 = (const float*)d_in[13];
  const float* Wv2 = (const float*)d_in[14], *bv2 = (const float*)d_in[15];
  const float* Wv3 = (const float*)d_in[16], *bv3 = (const float*)d_in[17];
  const float* Wq  = (const float*)d_in[18], *bq  = (const float*)d_in[19];
  const float* Wkv = (const float*)d_in[20], *bkv = (const float*)d_in[21];
  const float* Wb  = (const float*)d_in[22], *bb  = (const float*)d_in[23];
  const float* Wdz = (const float*)d_in[24], *bdz = (const float*)d_in[25];
  const float* Wo  = (const float*)d_in[26], *bo  = (const float*)d_in[27];
  const float* hw  = (const float*)d_in[28];
  float* out = (float*)d_out;

  const int nN = in_sizes[0] / CSd;
  const int nE = in_sizes[1] / CZd;

  // bump allocator over workspace
  char* base = (char*)d_ws;
  size_t off = 0;
  auto alloc = [&](size_t bytes) -> void* {
    void* p = base + off;
    off += (bytes + 255) & ~(size_t)255;
    return p;
  };
  auto nTiles = [](int K, int N) -> size_t {
    return (size_t)((K + 31) / 32) * ((N + 15) / 16) * 512;
  };

  _Float16* pWw1 = (_Float16*)alloc(nTiles(896, 192) * 2);
  _Float16* pWw2 = (_Float16*)alloc(nTiles(192, 192) * 2);
  _Float16* pWw3 = (_Float16*)alloc(nTiles(192, 12) * 2);
  _Float16* pWb  = (_Float16*)alloc(nTiles(128, 12) * 2);
  _Float16* pWv1 = (_Float16*)alloc(nTiles(512, 192) * 2);
  _Float16* pWv2 = (_Float16*)alloc(nTiles(192, 192) * 2);
  _Float16* pWv3 = (_Float16*)alloc(nTiles(192, 192) * 2);
  _Float16* pWdz = (_Float16*)alloc(nTiles(128, 32) * 2);
  _Float16* pWq  = (_Float16*)alloc(nTiles(384, 144) * 2);
  _Float16* pWkv = (_Float16*)alloc(nTiles(384, 432) * 2);
  _Float16* pWo  = (_Float16*)alloc(nTiles(960, 384) * 2);

  float* qpts  = (float*)alloc((size_t)nN * 144 * 4);
  float* kvpts = (float*)alloc((size_t)nN * 432 * 4);
  float* a_ex  = (float*)alloc((size_t)nE * NHD * 4);
  int*   amax  = (int*)alloc((size_t)nN * NHD * 4);
  size_t nz = (size_t)nN * (NHD + CHd + 288 + 384);  // denom + o + o_pt + o_pair
  float* zblock    = (float*)alloc(nz * 4);
  float* denom     = zblock;
  float* o_acc     = denom + (size_t)nN * NHD;
  float* opt_acc   = o_acc + (size_t)nN * CHd;
  float* opair_acc = opt_acc + (size_t)nN * 288;
  if (off > ws_size) return;  // insufficient workspace

  auto pack = [&](const float* W, _Float16* dst, int K, int N) {
    int KT = (K + 31) / 32, NT = (N + 15) / 16;
    size_t total = (size_t)KT * NT * 512;
    pack_weight<<<(unsigned)((total + 255) / 256), 256, 0, stream>>>(W, dst, K, N, KT, NT);
  };
  pack(Ww1, pWw1, 896, 192);
  pack(Ww2, pWw2, 192, 192);
  pack(Ww3, pWw3, 192, 12);
  pack(Wb,  pWb,  128, 12);
  pack(Wv1, pWv1, 512, 192);
  pack(Wv2, pWv2, 192, 192);
  pack(Wv3, pWv3, 192, 192);
  pack(Wdz, pWdz, 128, 32);
  pack(Wq,  pWq,  384, 144);
  pack(Wkv, pWkv, 384, 432);
  pack(Wo,  pWo,  960, 384);

  size_t na = (size_t)nN * NHD;
  size_t ninit = nz > na ? nz : na;
  init_ws<<<(unsigned)((ninit + 255) / 256), 256, 0, stream>>>(amax, zblock, nz, na);

  int nodeTiles16 = (nN + 15) / 16;
  int tiles32     = (nE + ET - 1) / ET;
  int nodeTiles32 = (nN + ET - 1) / ET;

  node_proj<<<nodeTiles16, 128, 0, stream>>>(s, rots, trans, pWq, bq, pWkv, bkv,
                                             qpts, kvpts, nN);

  edge_pass1<<<tiles32, 128, 0, stream>>>(s, z, eidx, qpts, kvpts, mask,
                                          pWw1, bw1, pWw2, bw2, pWw3, bw3, pWb, bb, hw,
                                          a_ex, amax, nE);

  edge_softmax<<<(nE * NHD + 255) / 256, 256, 0, stream>>>(a_ex, eidx, amax, denom, nE);

  edge_agg<<<tiles32, 128, 0, stream>>>(s, z, eidx, kvpts, a_ex, denom,
                                        pWv1, bv1, pWv2, bv2, pWv3, bv3, pWdz, bdz,
                                        o_acc, opt_acc, opair_acc, nE);

  node_out<<<nodeTiles32, 128, 0, stream>>>(o_acc, opt_acc, opair_acc, rots, trans,
                                            pWo, bo, out, nN);
}
